// TriangleMLP_36610301231756
// MI455X (gfx1250) — matrix operand order repaired and checked by reading, not compiled, not run on hardware
//
#include <hip/hip_runtime.h>
#include <stdint.h>

#define Hdim 2048
#define TM 16              // rows per workgroup
#define BUFDW (64 * 512)   // 128KB ping-pong buffer in dwords

typedef __attribute__((ext_vector_type(16))) __bf16 v16bf;
typedef __attribute__((ext_vector_type(8)))  float  v8f;

union FragU {
  uint32_t d[8];
  uint4    q[2];
  v16bf    v;
};

__device__ __forceinline__ uint16_t f2bf(float f) {
  uint32_t u = __float_as_uint(f);
  u += 0x7FFFu + ((u >> 16) & 1u);           // round-to-nearest-even
  return (uint16_t)(u >> 16);
}
__device__ __forceinline__ float bf2f(uint16_t h) {
  return __uint_as_float(((uint32_t)h) << 16);
}
__device__ __forceinline__ void split2(float x, uint16_t& hi, uint16_t& lo) {
  hi = f2bf(x);
  lo = f2bf(x - bf2f(hi));
}
__device__ __forceinline__ float leaky(float x) { return x > 0.f ? x : 0.01f * x; }

// ---------------------------------------------------------------------------
// Weight packing: masked + (optionally flipped) + transposed into the WMMA
// B-fragment lane layout, split into bf16 hi/lo blocks.
// Tile (kt,nt) = 512 dwords: [hi: 32 lanes x 8 dw][lo: 32 lanes x 8 dw].
// B 32x16 b16 layout: lane = group*16 + n%16 ; k_local = 2*(v&3) + p + 8*group + 16*(v>>2).
// mode: 0 = triu (keep W[n,k] if k>=n), 1 = flipped triu (W[n,H-1-k] if H-1-k>=n),
//       2 = dense (input layer, Kcols=64).
// ---------------------------------------------------------------------------
__global__ void pack_w(const float* __restrict__ W, uint32_t* __restrict__ out,
                       int Kcols, int mode) {
  int idx  = blockIdx.x * 256 + threadIdx.x;   // one thread per hi-dword
  int v    = idx & 7;
  int lane = (idx >> 3) & 31;
  int tile = idx >> 8;                          // tile = kt*128 + nt
  int nt   = tile & 127;
  int kt   = tile >> 7;
  if (kt * 32 >= Kcols) return;
  int n     = nt * 16 + (lane & 15);
  int group = lane >> 4;
  uint32_t hiD = 0, loD = 0;
#pragma unroll
  for (int p = 0; p < 2; ++p) {
    int kl = (v & 3) * 2 + p + 8 * group + 16 * (v >> 2);  // B-matrix k_local
    int k  = kt * 32 + kl;
    float w;
    if (mode == 2) {
      w = W[(size_t)n * Kcols + k];
    } else if (mode == 1) {
      int ks = Hdim - 1 - k;
      w = (ks >= n) ? W[(size_t)n * Hdim + ks] : 0.f;
    } else {
      w = (k >= n) ? W[(size_t)n * Hdim + k] : 0.f;
    }
    uint16_t hi, lo;
    split2(w, hi, lo);
    hiD |= ((uint32_t)hi) << (16 * p);
    loD |= ((uint32_t)lo) << (16 * p);
  }
  uint32_t base = (uint32_t)tile * 512u;
  out[base + lane * 8 + v]       = hiD;
  out[base + 256 + lane * 8 + v] = loD;
}

// ---------------------------------------------------------------------------
// One GEMM layer: in (LDS, A-frag hi/lo layout) x packed weights (global) ->
// bias + LeakyReLU + re-split -> outb (LDS, A-frag layout).
// Inlined per-layer so weight pointers trace directly to kernargs
// (=> global_load_b128, not flat_load) and B loads are double-buffered.
// ---------------------------------------------------------------------------
template <int KT>
__device__ __forceinline__ void do_layer(
    const uint32_t* __restrict__ in, uint32_t* __restrict__ outb,
    const uint32_t* __restrict__ wbase, const float* __restrict__ bv,
    int wave, int lane) {
  v8f acc[16];
#pragma unroll
  for (int j = 0; j < 16; ++j) acc[j] = {};

  for (int kt = 0; kt < KT; ++kt) {
    FragU ahi, alo;
    const uint32_t* ain = in + kt * 512 + lane * 8;
    ahi.q[0] = *(const uint4*)(ain);
    ahi.q[1] = *(const uint4*)(ain + 4);
    alo.q[0] = *(const uint4*)(ain + 256);
    alo.q[1] = *(const uint4*)(ain + 260);

    const uint32_t* tb = wbase + (size_t)(kt * 128 + wave * 16) * 512 + lane * 8;
    FragU bh[2], bl[2];
    bh[0].q[0] = *(const uint4*)(tb);
    bh[0].q[1] = *(const uint4*)(tb + 4);
    bl[0].q[0] = *(const uint4*)(tb + 256);
    bl[0].q[1] = *(const uint4*)(tb + 260);

#pragma unroll
    for (int j = 0; j < 16; ++j) {
      const int cur = j & 1, nxt = cur ^ 1;
      if (j < 15) {   // prefetch B fragments for tile j+1 while computing j
        const uint32_t* tn = tb + (j + 1) * 512;
        bh[nxt].q[0] = *(const uint4*)(tn);
        bh[nxt].q[1] = *(const uint4*)(tn + 4);
        bl[nxt].q[0] = *(const uint4*)(tn + 256);
        bl[nxt].q[1] = *(const uint4*)(tn + 260);
      }
      // bf16x3 split-precision: A*B ~= Ahi*Bhi + Ahi*Blo + Alo*Bhi (f32 accum)
      acc[j] = __builtin_amdgcn_wmma_f32_16x16x32_bf16(
          false, ahi.v, false, bh[cur].v, (short)0, acc[j], false, false);
      acc[j] = __builtin_amdgcn_wmma_f32_16x16x32_bf16(
          false, ahi.v, false, bl[cur].v, (short)0, acc[j], false, false);
      acc[j] = __builtin_amdgcn_wmma_f32_16x16x32_bf16(
          false, alo.v, false, bh[cur].v, (short)0, acc[j], false, false);
    }
  }

  // ---- bias + LeakyReLU + re-split, write into the other LDS buffer ----
#pragma unroll
  for (int j = 0; j < 16; ++j) {
    int nt  = wave * 16 + j;
    int col = nt * 16 + (lane & 15);       // D frag: N = lane%16
    float b = bv[col];
    int kt    = col >> 5;                  // col becomes next layer's k
    int kl    = col & 31;
    int pbit  = kl & 1;
    int vv    = ((kl >> 4) & 1) * 4 + ((kl >> 1) & 3);
    int group = (kl >> 3) & 1;
    uint16_t* hb = (uint16_t*)(outb + kt * 512);
    uint16_t* lb = (uint16_t*)(outb + kt * 512 + 256);
#pragma unroll
    for (int r = 0; r < 8; ++r) {
      int row = r + 8 * (lane >> 4);       // D frag: M = r + 8*(lane/16)
      float y = leaky(acc[j][r] + b);
      uint16_t hi, lo;
      split2(y, hi, lo);
      int lp = group * 16 + row;
      hb[(lp * 8 + vv) * 2 + pbit] = hi;
      lb[(lp * 8 + vv) * 2 + pbit] = lo;
    }
  }
}

// ---------------------------------------------------------------------------
// Fused MLP. Activations live in LDS in A-fragment layout (hi/lo bf16 pairs),
// ping-ponging between two 128KB buffers. A 16x32 b16 layout:
//   lane = group*16 + m ; k_local = 2*(v&3) + p + 8*group + 16*(v>>2).
// ---------------------------------------------------------------------------
__global__ void __launch_bounds__(256, 1) mlp_fused(
    const float* __restrict__ p,
    const float* __restrict__ bias0, const float* __restrict__ bias1,
    const float* __restrict__ bias2, const float* __restrict__ bias3,
    const float* __restrict__ out_w, const float* __restrict__ out_b,
    const uint32_t* __restrict__ w0p, const uint32_t* __restrict__ w1p,
    const uint32_t* __restrict__ w2p, const uint32_t* __restrict__ w3p,
    float* __restrict__ out) {
  extern __shared__ uint32_t smem[];
  const int tid  = threadIdx.x;
  const int wave = tid >> 5;
  const int lane = tid & 31;
  const int row0 = blockIdx.x * TM;

  // ---- Stage input rows p[16 x 64] into buffer 0 in A-fragment layout ----
  for (int i = tid; i < 512; i += 256) {          // i indexes hi-dwords (2 k-tiles)
    int kt = i >> 8, rem = i & 255;
    int lp = rem >> 3, v = rem & 7;
    int group = lp >> 4, row = lp & 15;
    int k0 = kt * 32 + ((v & 3) * 2 + 8 * group + 16 * (v >> 2));
    float f0 = p[(size_t)(row0 + row) * 64 + k0];
    float f1 = p[(size_t)(row0 + row) * 64 + k0 + 1];
    uint16_t h0, l0, h1, l1;
    split2(f0, h0, l0);
    split2(f1, h1, l1);
    smem[kt * 512 + lp * 8 + v]       = (uint32_t)h0 | ((uint32_t)h1 << 16);
    smem[kt * 512 + 256 + lp * 8 + v] = (uint32_t)l0 | ((uint32_t)l1 << 16);
  }
  __syncthreads();

  // Layers unrolled explicitly: weight pointers stay kernarg-derived.
  do_layer<2>(smem, smem + BUFDW, w0p, bias0, wave, lane);
  __syncthreads();
  do_layer<64>(smem + BUFDW, smem, w1p, bias1, wave, lane);
  __syncthreads();
  do_layer<64>(smem, smem + BUFDW, w2p, bias2, wave, lane);
  __syncthreads();
  do_layer<64>(smem + BUFDW, smem, w3p, bias3, wave, lane);
  __syncthreads();

  // ---- Output layer: [16 x 2048] @ out_w^T (3 cols), exact fp32 dots ----
  const uint32_t* xin = smem;                       // layer-3 output buffer
  for (int rr = 0; rr < 2; ++rr) {
    int row = wave * 2 + rr;
    float s0 = 0.f, s1 = 0.f, s2 = 0.f;
    for (int k = lane; k < Hdim; k += 32) {
      int kt = k >> 5, kl = k & 31;
      int pbit  = kl & 1;
      int vv    = ((kl >> 4) & 1) * 4 + ((kl >> 1) & 3);
      int group = (kl >> 3) & 1;
      int lp    = group * 16 + row;
      const uint16_t* hb = (const uint16_t*)(xin + kt * 512);
      const uint16_t* lb = (const uint16_t*)(xin + kt * 512 + 256);
      float x = bf2f(hb[(lp * 8 + vv) * 2 + pbit]) +
                bf2f(lb[(lp * 8 + vv) * 2 + pbit]);
      s0 += x * out_w[k];
      s1 += x * out_w[Hdim + k];
      s2 += x * out_w[2 * Hdim + k];
    }
#pragma unroll
    for (int off = 16; off > 0; off >>= 1) {
      s0 += __shfl_xor(s0, off, 32);
      s1 += __shfl_xor(s1, off, 32);
      s2 += __shfl_xor(s2, off, 32);
    }
    if (lane == 0) {
      size_t o = (size_t)(row0 + row) * 3;
      out[o + 0] = s0 + out_b[0];
      out[o + 1] = s1 + out_b[1];
      out[o + 2] = s2 + out_b[2];
    }
  }
}

extern "C" void kernel_launch(void* const* d_in, const int* in_sizes, int n_in,
                              void* d_out, int out_size, void* d_ws, size_t ws_size,
                              hipStream_t stream) {
  const float* p     = (const float*)d_in[0];
  const float* initw = (const float*)d_in[1];
  const float* initb = (const float*)d_in[2];
  const float* w1 = (const float*)d_in[3];
  const float* b1 = (const float*)d_in[4];
  const float* w2 = (const float*)d_in[5];
  const float* b2 = (const float*)d_in[6];
  const float* w3 = (const float*)d_in[7];
  const float* b3 = (const float*)d_in[8];
  const float* ow = (const float*)d_in[9];
  const float* ob = (const float*)d_in[10];

  uint32_t* ws  = (uint32_t*)d_ws;
  uint32_t* w0p = ws;                        //  256 tiles * 512 dw (0.5 MB)
  uint32_t* w1p = w0p + 256u * 512u;         // 8192 tiles * 512 dw (16 MB)
  uint32_t* w2p = w1p + 8192u * 512u;
  uint32_t* w3p = w2p + 8192u * 512u;        // total ~48.5 MB, L2-resident

  pack_w<<<256,  256, 0, stream>>>(initw, w0p, 64, 2);
  pack_w<<<8192, 256, 0, stream>>>(w1, w1p, 2048, 0);
  pack_w<<<8192, 256, 0, stream>>>(w2, w2p, 2048, 0);
  pack_w<<<8192, 256, 0, stream>>>(w3, w3p, 2048, 1);  // fold column flip in

  int nrows = in_sizes[0] / 64;              // 32768
  size_t lds = 2u * (size_t)BUFDW * sizeof(uint32_t);  // 256 KB dynamic LDS
  mlp_fused<<<nrows / TM, 256, lds, stream>>>(
      p, initb, b1, b2, b3, ow, ob, w0p, w1p, w2p, w3p, (float*)d_out);
}
